// _EEGMiner_16638703305037
// MI455X (gfx1250) — compile-verified, matrix operand order checked by reading, not run on hardware
//
#include <hip/hip_runtime.h>
#include <cmath>

#define DEV __device__ __forceinline__

constexpr int   B      = 64;
constexpr int   C      = 64;
constexpr int   T      = 4096;
constexpr int   NF     = 2;
constexpr int   NOUT   = 4;
constexpr int   F_BINS = T / 2 + 1;        // 2049
constexpr int   NPAIR  = C * (C - 1) / 2;  // 2016
constexpr int   N_FEAT = NF * NPAIR;       // 4032
constexpr float NYQ    = 128.0f;
constexpr float PI_F   = 3.14159265358979323846f;

typedef __attribute__((ext_vector_type(16))) _Float16 v16h;
typedef __attribute__((ext_vector_type(8)))  _Float16 v8h;
typedef __attribute__((ext_vector_type(8)))  float    v8f;

// ---------------------------------------------------------------------------
// K0: construct the NF generalized-Gaussian filters on-device.
//   n_range[k] = k/2048, phase grid = k/8.
// ---------------------------------------------------------------------------
__global__ void __launch_bounds__(256)
k_filters(const float* __restrict__ f_mean, const float* __restrict__ bw_in,
          const float* __restrict__ shp,    const float* __restrict__ gdl,
          float* __restrict__ filt_re, float* __restrict__ filt_im) {
  const int f   = blockIdx.x;
  const int tid = threadIdx.x;
  __shared__ float red[256];

  const float fm = fminf(fmaxf(f_mean[f], 1.0f / NYQ), 45.0f / NYQ);
  const float bw = fminf(fmaxf(bw_in[f], 1.0f / NYQ), 1.0f);
  const float p  = fminf(fmaxf(shp[f], 2.0f), 3.0f) * 8.0f - 14.0f;
  const float gd = gdl[f];
  const float scale = bw / (2.0f * powf(0.69314718055994530942f, 1.0f / p));

  float mg[9];
  float lmax = 0.0f;
#pragma unroll
  for (int s = 0; s < 9; ++s) {
    const int k = tid + s * 256;
    float m = -1.0f;
    if (k < F_BINS) {
      const float nr = (float)k * (1.0f / 2048.0f);
      m = expf(-powf((fabsf(nr - fm) + 1e-8f) / scale, p));
      lmax = fmaxf(lmax, m);
    }
    mg[s] = m;
  }
  red[tid] = lmax;
  __syncthreads();
  for (int off = 128; off > 0; off >>= 1) {
    if (tid < off) red[tid] = fmaxf(red[tid], red[tid + off]);
    __syncthreads();
  }
  const float inv_max = 1.0f / red[0];

#pragma unroll
  for (int s = 0; s < 9; ++s) {
    const int k = tid + s * 256;
    if (k < F_BINS) {
      const float mag = mg[s] * inv_max;
      const float pha = -gd * ((float)k * 0.125f) * PI_F;
      float sn, cs;
      __sincosf(pha, &sn, &cs);
      filt_re[f * F_BINS + k] = mag * cs;
      filt_im[f * F_BINS + k] = mag * sn;
    }
  }
}

// ---------------------------------------------------------------------------
// In-LDS 4096-point radix-2 complex FFT (sign = -1 fwd, +1 inv, unscaled).
// ---------------------------------------------------------------------------
DEV void fft4096(float* re, float* im, int tid, float sign) {
  __syncthreads();
  for (int i = tid; i < T; i += 256) {
    const int j = (int)(__brev((unsigned)i) >> 20);  // 12-bit reverse
    if (j > i) {
      float tr = re[i]; re[i] = re[j]; re[j] = tr;
      float ti = im[i]; im[i] = im[j]; im[j] = ti;
    }
  }
  __syncthreads();
  for (int s = 0; s < 12; ++s) {
    const int half = 1 << s;
    for (int q = tid; q < T / 2; q += 256) {
      const int j    = q & (half - 1);
      const int base = ((q >> s) << (s + 1)) + j;
      const int part = base + half;
      const float ang = sign * (2.0f * PI_F) * (float)j / (float)(half << 1);
      float sn, cs;
      __sincosf(ang, &sn, &cs);
      const float pr = re[part], pi = im[part];
      const float tr = pr * cs - pi * sn;
      const float ti = pr * sn + pi * cs;
      const float br = re[base], bi = im[base];
      re[part] = br - tr; im[part] = bi - ti;
      re[base] = br + tr; im[base] = bi + ti;
    }
    __syncthreads();
  }
}

// ---------------------------------------------------------------------------
// K1: per (b,c) — rfft, then per filter: component-wise filter + hilbert
// scaling, inverse FFT, unit-phase vectors -> f16 streams Ur/Ui.
// ---------------------------------------------------------------------------
__global__ void __launch_bounds__(256)
k_fft_filter_hilbert(const float* __restrict__ x,
                     const float* __restrict__ filt_re,
                     const float* __restrict__ filt_im,
                     _Float16* __restrict__ Ur, _Float16* __restrict__ Ui) {
  __shared__ float re[T];
  __shared__ float im[T];
  const int tid = threadIdx.x;
  const int b = blockIdx.x / C;
  const int c = blockIdx.x % C;

  const float* xp = x + (size_t)blockIdx.x * T;
  for (int i = tid; i < T; i += 256) { re[i] = xp[i]; im[i] = 0.0f; }

  fft4096(re, im, tid, -1.0f);  // forward: bins 0..2048 == rfft(x)

  float sxr[16], sxi[16];       // register stash of the spectrum
#pragma unroll
  for (int s = 0; s < 16; ++s) {
    const int k = tid + s * 256;
    sxr[s] = re[k];
    sxi[s] = im[k];
  }

  for (int f = 0; f < NF; ++f) {
    __syncthreads();
#pragma unroll
    for (int s = 0; s < 16; ++s) {
      const int k = tid + s * 256;
      if (k < F_BINS) {
        const float dcs = (k == 0) ? 1.0f : 2.0f;  // hilbert: 2x all, 1x DC
        re[k] = sxr[s] * filt_re[f * F_BINS + k] * dcs;  // component-wise
        im[k] = sxi[s] * filt_im[f * F_BINS + k] * dcs;  // (not complex mul)
      } else {
        re[k] = 0.0f; im[k] = 0.0f;
      }
    }
    fft4096(re, im, tid, 1.0f);  // inverse (scale 1/T applied below)

    _Float16* ur = Ur + (size_t)((b * NF + f) * C + c) * T;
    _Float16* ui = Ui + (size_t)((b * NF + f) * C + c) * T;
#pragma unroll
    for (int s = 0; s < 16; ++s) {
      const int k = tid + s * 256;
      const float ar = re[k] * (1.0f / (float)T);
      const float ai = im[k] * (1.0f / (float)T);
      const float inv_amp = rsqrtf(ar * ar + ai * ai + 1e-6f);
      ur[k] = (_Float16)(ar * inv_amp);
      ui[k] = (_Float16)(ai * inv_amp);
    }
  }
}

// ---------------------------------------------------------------------------
// K2: per (b,f) — three 64x64x4096 grams via v_wmma_f32_16x16x32_f16,
// then PLV upper-triangle features.
// A fragment: lane<16 -> row=lane, K={0..7,16..23}; lane>=16 -> K={8..15,24..31}.
// Loading the n-tile rows with this same pattern yields the (mirrored) B
// operand, i.e. Uᵀ, so D = U_m · U_nᵀ without any transpose instructions.
// ---------------------------------------------------------------------------
DEV v16h load_frag(const _Float16* __restrict__ base, int row0, int kk, int lane) {
  const int r  = lane & 15;
  const int kg = (lane >> 4) << 3;  // 0 or 8
  const _Float16* p = base + (size_t)(row0 + r) * T + (kk + kg);
  const v8h lo = *(const v8h*)(p);
  const v8h hi = *(const v8h*)(p + 16);
  return __builtin_shufflevector(lo, hi, 0, 1, 2, 3, 4, 5, 6, 7,
                                         8, 9, 10, 11, 12, 13, 14, 15);
}

DEV v8f wmma_f16(v16h a, v16h b, v8f c) {
  return __builtin_amdgcn_wmma_f32_16x16x32_f16(
      /*neg_a=*/false, a, /*neg_b=*/false, b,
      /*c_mod=*/(short)0, c, /*reuse_a=*/false, /*reuse_b=*/false);
}

__global__ void __launch_bounds__(256)
k_gram_plv(const _Float16* __restrict__ Ur, const _Float16* __restrict__ Ui,
           float* __restrict__ feats) {
  __shared__ float g_re[C][C + 1];  // rr + ii   (pad: conflict-free transpose)
  __shared__ float g_ri[C][C + 1];  // ri
  const int bf   = blockIdx.x;      // b*NF + f
  const int b    = bf / NF;
  const int f    = bf % NF;
  const int tid  = threadIdx.x;
  const int lane = tid & 31;
  const int wave = tid >> 5;        // 8 waves x 2 tiles = 4x4 tile grid

  const _Float16* ur = Ur + (size_t)bf * C * T;
  const _Float16* ui = Ui + (size_t)bf * C * T;

  const int t0  = wave * 2;
  const int mt  = t0 >> 2;
  const int nt0 = t0 & 3;
  const int mb  = mt * 16, nb0 = nt0 * 16, nb1 = nb0 + 16;

  v8f rr0 = {}, ii0 = {}, ri0 = {};
  v8f rr1 = {}, ii1 = {}, ri1 = {};

  for (int kk = 0; kk < T; kk += 32) {
    const v16h ar  = load_frag(ur, mb,  kk, lane);
    const v16h ai  = load_frag(ui, mb,  kk, lane);
    const v16h br0 = load_frag(ur, nb0, kk, lane);
    const v16h bi0 = load_frag(ui, nb0, kk, lane);
    const v16h br1 = load_frag(ur, nb1, kk, lane);
    const v16h bi1 = load_frag(ui, nb1, kk, lane);
    rr0 = wmma_f16(ar, br0, rr0);
    ii0 = wmma_f16(ai, bi0, ii0);
    ri0 = wmma_f16(ar, bi0, ri0);
    rr1 = wmma_f16(ar, br1, rr1);
    ii1 = wmma_f16(ai, bi1, ii1);
    ri1 = wmma_f16(ar, bi1, ri1);
  }

  // C/D layout: lane<16 -> N=lane, M=r ; lane>=16 -> N=lane-16, M=8+r
  const int n  = lane & 15;
  const int mo = (lane >> 4) * 8;
#pragma unroll
  for (int r = 0; r < 8; ++r) {
    g_re[mb + mo + r][nb0 + n] = rr0[r] + ii0[r];
    g_ri[mb + mo + r][nb0 + n] = ri0[r];
    g_re[mb + mo + r][nb1 + n] = rr1[r] + ii1[r];
    g_ri[mb + mo + r][nb1 + n] = ri1[r];
  }
  __syncthreads();

  // PLV over upper triangle: corr_im[i][j] = ri[i][j] - ri[j][i]
  for (int p = tid; p < NPAIR; p += 256) {
    int i = 0, rem = p, span = C - 1;
    while (rem >= span) { rem -= span; ++i; --span; }
    const int j = i + 1 + rem;
    const float cre = g_re[i][j];
    const float cim = g_ri[i][j] - g_ri[j][i];
    const float plv = sqrtf(cre * cre + cim * cim + 1e-6f) * (1.0f / (float)T);
    feats[(size_t)b * N_FEAT + (size_t)p * NF + f] = plv;
  }
}

// ---------------------------------------------------------------------------
// K3: BatchNorm1d (affine=False, training): biased batch stats, eps=1e-5.
// One thread owns one feature column (B=64 rows).
// ---------------------------------------------------------------------------
__global__ void __launch_bounds__(256)
k_bnorm(float* __restrict__ feats) {
  const int j = blockIdx.x * blockDim.x + threadIdx.x;
  if (j >= N_FEAT) return;
  float mu = 0.0f;
  for (int b = 0; b < B; ++b) mu += feats[(size_t)b * N_FEAT + j];
  mu *= (1.0f / (float)B);
  float var = 0.0f;
  for (int b = 0; b < B; ++b) {
    const float d = feats[(size_t)b * N_FEAT + j] - mu;
    var += d * d;
  }
  var *= (1.0f / (float)B);
  const float inv = rsqrtf(var + 1e-5f);
  for (int b = 0; b < B; ++b)
    feats[(size_t)b * N_FEAT + j] = (feats[(size_t)b * N_FEAT + j] - mu) * inv;
}

// ---------------------------------------------------------------------------
// K4: out[b,o] = feats_norm[b,:] . lin_w[o,:] + lin_b[o]   (64x4 outputs)
// ---------------------------------------------------------------------------
__global__ void __launch_bounds__(256)
k_linear(const float* __restrict__ feats, const float* __restrict__ w,
         const float* __restrict__ bias, float* __restrict__ out) {
  const int t = threadIdx.x;   // 256 = B*NOUT
  const int b = t >> 2, o = t & 3;
  float acc = bias[o];
  const float* fr = feats + (size_t)b * N_FEAT;
  const float* wr = w + (size_t)o * N_FEAT;
  for (int j = 0; j < N_FEAT; ++j) acc += fr[j] * wr[j];
  out[b * NOUT + o] = acc;
}

// ---------------------------------------------------------------------------
extern "C" void kernel_launch(void* const* d_in, const int* in_sizes, int n_in,
                              void* d_out, int out_size, void* d_ws, size_t ws_size,
                              hipStream_t stream) {
  const float* x     = (const float*)d_in[0];
  const float* fmean = (const float*)d_in[1];
  const float* bwid  = (const float*)d_in[2];
  const float* shp   = (const float*)d_in[3];
  const float* gdl   = (const float*)d_in[4];
  const float* lw    = (const float*)d_in[5];
  const float* lb    = (const float*)d_in[6];
  float* out = (float*)d_out;

  // Workspace layout:
  //   feats   : B*N_FEAT floats (~1 MB)
  //   filt_re : NF*F_BINS floats
  //   filt_im : NF*F_BINS floats
  //   Ur, Ui  : B*NF*C*T halves each (~64 MB each), 16B-aligned
  float* feats   = (float*)d_ws;
  float* filt_re = feats + (size_t)B * N_FEAT;
  float* filt_im = filt_re + (size_t)NF * F_BINS;
  size_t fbytes = ((size_t)B * N_FEAT + 2u * (size_t)NF * F_BINS) * sizeof(float);
  fbytes = (fbytes + 31) & ~(size_t)31;
  _Float16* Ur = (_Float16*)((char*)d_ws + fbytes);
  _Float16* Ui = Ur + (size_t)B * NF * C * T;

  k_filters<<<NF, 256, 0, stream>>>(fmean, bwid, shp, gdl, filt_re, filt_im);
  k_fft_filter_hilbert<<<B * C, 256, 0, stream>>>(x, filt_re, filt_im, Ur, Ui);
  k_gram_plv<<<B * NF, 256, 0, stream>>>(Ur, Ui, feats);
  k_bnorm<<<(N_FEAT + 255) / 256, 256, 0, stream>>>(feats);
  k_linear<<<1, 256, 0, stream>>>(feats, lw, lb, out);
}
